// WindowMultiHeadAttention_4260607558254
// MI455X (gfx1250) — matrix operand order repaired and checked by reading, not compiled, not run on hardware
//
#include <hip/hip_runtime.h>
#include <hip/hip_bf16.h>

typedef __attribute__((ext_vector_type(16))) _Float16 v16h;
typedef __attribute__((ext_vector_type(8)))  float    v8f;

#define N_TOK 49
#define NPAD  64
#define NH    4
#define CIN   128
#define HW    56
#define SCALE_QK 0.17677669529663687f   // 1/sqrt(32)

// ---------------- LDS layout (bytes) ----------------
#define LW_STRIDE 136    // halfs per row, 128-col f16 buffers (+8 pad vs 256B bank period)
#define VT_STRIDE 72
#define SC_STRIDE 68     // floats per row
#define PB_STRIDE 72

#define WSZ    (128 * LW_STRIDE * 2)        // 34816 per weight
#define W_ALL  (4 * WSZ)                    // 139264: all four weights, padded f16 image
#define OFF_WQ 0
#define OFF_WK (OFF_WQ + WSZ)
#define OFF_WV (OFF_WK + WSZ)
#define OFF_WO (OFF_WV + WSZ)
#define XSZ    (NPAD * LW_STRIDE * 2)       // 17408
#define OFF_U  (OFF_WO + WSZ)               // 139264: union region
#define OFF_XQ (OFF_U)                      // phase 1-2: window inputs
#define OFF_XK (OFF_XQ + XSZ)
#define OFF_XV (OFF_XK + XSZ)
#define OFF_SC (OFF_U)                      // phase 3-4: fp32 scores 4*64*68*4 = 69632
#define OFF_PB (OFF_U + 69632)              // f16 probs 4*64*72*2 = 36864
#define OFF_QB (OFF_U + 106496)             // 245760: projected Q
#define OFF_KB (OFF_QB + XSZ)               // projected K
#define OFF_AO (OFF_QB)                     // attn-out overlays dead Q
#define OFF_VT (OFF_KB + XSZ)               // 280576: V transposed [c][j], 128*72*2
#define SMEM_BYTES (OFF_VT + 128 * VT_STRIDE * 2)   // 299008 <= 320KB/WGP

// ---- CDNA5 async global->LDS copy (ASYNCcnt-tracked, no VGPR round trip) ----
__device__ __forceinline__ void async_ld_b128(void* lds_ptr, const void* gptr) {
    uint32_t l = (uint32_t)(uintptr_t)lds_ptr;   // low 32 bits = wave-relative LDS offset
    asm volatile("global_load_async_to_lds_b128 %0, %1, off"
                 :: "v"(l), "v"(gptr) : "memory");
}
__device__ __forceinline__ void wait_async0() {
    asm volatile("s_wait_asynccnt 0x0" ::: "memory");
}

// ---- WMMA fragment loaders (wave32, f16, from row-major [row][K] LDS) ----
// A 16x32: lanes 0-15 hold K pairs {0..7,16..23}, lanes 16-31 shifted +8.
__device__ __forceinline__ v16h load_frag_a(const _Float16* base, int tile, int stride, int kbase) {
    const int lane = threadIdx.x & 31;
    const uint32_t* p = (const uint32_t*)(base + (tile * 16 + (lane & 15)) * stride
                                               + kbase + ((lane & 16) ? 8 : 0));
    union { v16h h; uint32_t u[8]; } f;
    f.u[0] = p[0]; f.u[1] = p[1]; f.u[2] = p[2];  f.u[3] = p[3];
    f.u[4] = p[8]; f.u[5] = p[9]; f.u[6] = p[10]; f.u[7] = p[11];
    return f.h;
}
// B 32x16: lane holds column N=lane&15; lanes 0-15: K {0..7,16..23}, lanes 16-31 shifted +8.
__device__ __forceinline__ v16h load_frag_b(const _Float16* base, int tile, int stride, int kbase) {
    const int lane = threadIdx.x & 31;
    const uint32_t* p = (const uint32_t*)(base + (tile * 16 + (lane & 15)) * stride
                                               + kbase + ((lane & 16) ? 8 : 0));
    union { v16h h; uint32_t u[8]; } f;
#pragma unroll
    for (int i = 0; i < 4; ++i) { f.u[i] = p[i]; f.u[i + 4] = p[i + 8]; }
    return f.h;
}

__device__ __forceinline__ v8f wmma_f16(v16h a, v16h b, v8f c) {
    return __builtin_amdgcn_wmma_f32_16x16x32_f16(false, a, false, b, (short)0, c, false, false);
}

// ---------------- Prep kernel: fp32 weights -> padded f16 image in workspace (once) --------
__global__ __launch_bounds__(256) void convert_weights_f16(
    const float* __restrict__ Wq, const float* __restrict__ Wk,
    const float* __restrict__ Wv, const float* __restrict__ Wo,
    _Float16* __restrict__ dst)
{
    int e = blockIdx.x * 256 + threadIdx.x;          // over 4*128*136 padded halfs
    if (e >= 4 * 128 * LW_STRIDE) return;
    const int wi  = e / (128 * LW_STRIDE);
    const int rem = e - wi * (128 * LW_STRIDE);
    const int r   = rem / LW_STRIDE;
    const int c   = rem - r * LW_STRIDE;
    const float* src = (wi == 0) ? Wq : (wi == 1) ? Wk : (wi == 2) ? Wv : Wo;
    dst[e] = (c < 128) ? (_Float16)src[r * 128 + c] : (_Float16)0.f;
}

__global__ __launch_bounds__(256) void swin_window_attn_fused(
    const float* __restrict__ q_in, const float* __restrict__ k_in, const float* __restrict__ v_in,
    const _Float16* __restrict__ w16,     // prepped padded f16 weights (WQ|WK|WV|WO)
    const float* __restrict__ bq, const float* __restrict__ bk,
    const float* __restrict__ bv, const float* __restrict__ bo,
    const float* __restrict__ rpb,
    float* __restrict__ out_spatial, float* __restrict__ out_ctx)
{
    extern __shared__ char smem[];
    const int tid  = threadIdx.x;
    const int lane = tid & 31;
    const int wave = tid >> 5;
    const int mt   = wave & 3;       // M-tile owned by this wave
    const int ng   = wave >> 2;      // N-tile group (0 or 1)
    const int hi8  = (lane & 16) ? 8 : 0;
    const int nl   = lane & 15;

    const int w    = blockIdx.x;     // window id 0..4095
    const int bimg = w >> 6;
    const int win  = w & 63;
    const int y0   = (win >> 3) * 7;
    const int x0   = (win & 7) * 7;
    const int base = bimg * CIN * HW * HW;   // < 2^31

    _Float16* sWq = (_Float16*)(smem + OFF_WQ);
    _Float16* sWk = (_Float16*)(smem + OFF_WK);
    _Float16* sWv = (_Float16*)(smem + OFF_WV);
    _Float16* sWo = (_Float16*)(smem + OFF_WO);
    _Float16* sQ  = (_Float16*)(smem + OFF_QB);
    _Float16* sK  = (_Float16*)(smem + OFF_KB);
    _Float16* sVT = (_Float16*)(smem + OFF_VT);

    // -------- Phase 1a: kick off async copy of prepped f16 weights into LDS ---------------
    // 139264 B in 16B chunks: 8704 chunks = 34 per thread; ASYNCcnt-tracked, overlaps 1b.
    {
        const char* gsrc = (const char*)w16;
#pragma unroll
        for (int it = 0; it < W_ALL / (256 * 16); ++it) {
            const int ofs = (it * 256 + tid) * 16;
            async_ld_b128(smem + ofs, gsrc + ofs);
        }
    }

    // -------- Phase 1b: stage window inputs fp32 -> f16 (VALU work overlaps the DMA) ------
    {
        const float* xsrc[3] = {q_in, k_in, v_in};
        _Float16*    xdst[3] = {(_Float16*)(smem + OFF_XQ),
                                (_Float16*)(smem + OFF_XK),
                                (_Float16*)(smem + OFF_XV)};
#pragma unroll
        for (int p = 0; p < 3; ++p) {
            const float* src = xsrc[p];
            _Float16*    dst = xdst[p];
            for (int e = tid; e < N_TOK * CIN; e += 256) {
                int c = e / 49, t = e - c * 49;
                int ty = t / 7, tx = t - ty * 7;
                dst[t * LW_STRIDE + c] =
                    (_Float16)src[base + (c * HW + (y0 + ty)) * HW + (x0 + tx)];
            }
            for (int e = tid; e < (NPAD - N_TOK) * CIN; e += 256) {
                int t = N_TOK + (e >> 7), c = e & 127;
                dst[t * LW_STRIDE + c] = (_Float16)0.f;
            }
        }
    }
    wait_async0();
    __syncthreads();

    // ---------------- Phase 2: QKV projections (64x128 = x[64x128] @ W^T[128x128]) --------
    {
        _Float16* xb[3] = {(_Float16*)(smem + OFF_XQ), (_Float16*)(smem + OFF_XK),
                           (_Float16*)(smem + OFF_XV)};
        _Float16* wb[3] = {sWq, sWk, sWv};
        const float* bb[3] = {bq, bk, bv};
#pragma unroll
        for (int p = 0; p < 3; ++p) {
#pragma unroll
            for (int n2 = 0; n2 < 4; ++n2) {
                const int nt = ng * 4 + n2;
                v8f acc = {};
#pragma unroll
                for (int kc = 0; kc < 4; ++kc) {
                    v16h a  = load_frag_a(xb[p], mt, LW_STRIDE, kc * 32);
                    v16h bm = load_frag_b(wb[p], nt, LW_STRIDE, kc * 32);
                    acc = wmma_f16(a, bm, acc);
                }
                const int c  = nt * 16 + nl;
                const float bias = bb[p][c];
                if (p < 2) {
                    _Float16* dst = (p == 0) ? sQ : sK;
#pragma unroll
                    for (int r = 0; r < 8; ++r)
                        dst[(mt * 16 + r + hi8) * LW_STRIDE + c] = (_Float16)(acc[r] + bias);
                } else {   // store V transposed: VT[c][token]
#pragma unroll
                    for (int r = 0; r < 8; ++r)
                        sVT[c * VT_STRIDE + (mt * 16 + r + hi8)] = (_Float16)(acc[r] + bias);
                }
            }
        }
    }
    __syncthreads();   // X region now dead -> reused for scores/probs

    // ---------------- Phase 3: scores = (q k^T)*scale + rel-pos bias ----------------------
    {
        const int h    = wave >> 1;
        const int half = wave & 1;
        float* sc = (float*)(smem + OFF_SC) + h * (NPAD * SC_STRIDE);
#pragma unroll
        for (int m2 = 0; m2 < 4; ++m2) {
#pragma unroll
            for (int n2 = 0; n2 < 2; ++n2) {
                const int nt = half * 2 + n2;
                v8f acc = {};
                v16h a  = load_frag_a(sQ, m2, LW_STRIDE, h * 32);
                v16h bm = load_frag_b(sK, nt, LW_STRIDE, h * 32);
                acc = wmma_f16(a, bm, acc);
                const int j = nt * 16 + nl;
#pragma unroll
                for (int r = 0; r < 8; ++r) {
                    const int i = m2 * 16 + r + hi8;
                    float v = -1e30f;
                    if (i < N_TOK && j < N_TOK) {
                        const int idx = (i / 7 - j / 7 + 6) * 13 + (i % 7 - j % 7 + 6);
                        v = acc[r] * SCALE_QK + rpb[idx * NH + h];
                    }
                    sc[i * SC_STRIDE + j] = v;
                }
            }
        }
    }
    __syncthreads();

    // ---------------- Phase 4: softmax rows, emit ctx (fp32) + probs (f16) ----------------
    {
        float*    SC = (float*)(smem + OFF_SC);
        _Float16* PB = (_Float16*)(smem + OFF_PB);
        if (tid < NH * N_TOK) {              // 196 live rows
            const int h = tid / 49, i = tid - h * 49;
            float* row = SC + h * NPAD * SC_STRIDE + i * SC_STRIDE;
            float mx = -1e30f;
            for (int j = 0; j < N_TOK; ++j) mx = fmaxf(mx, row[j]);
            float s = 0.f;
            for (int j = 0; j < N_TOK; ++j) { float e = __expf(row[j] - mx); row[j] = e; s += e; }
            const float inv = 1.f / s;
            _Float16* prow = PB + h * NPAD * PB_STRIDE + i * PB_STRIDE;
            for (int j = 0; j < N_TOK; ++j) {
                float p = row[j] * inv;
                row[j]  = p;
                prow[j] = (_Float16)p;
            }
            for (int j = N_TOK; j < NPAD; ++j) prow[j] = (_Float16)0.f;
        } else {                             // threads 196..255 zero the 60 padded rows
            const int r2 = tid - NH * N_TOK;
            const int h = r2 / 15, i = N_TOK + (r2 - h * 15);
            _Float16* prow = PB + h * NPAD * PB_STRIDE + i * PB_STRIDE;
            for (int j = 0; j < NPAD; ++j) prow[j] = (_Float16)0.f;
        }
        __syncthreads();
        float* ctx = out_ctx + (long)w * (NH * N_TOK * N_TOK);
        for (int e = tid; e < NH * N_TOK * N_TOK; e += 256) {
            const int h = e / 2401, rem = e - h * 2401;
            const int i = rem / 49, j = rem - i * 49;
            ctx[e] = SC[h * NPAD * SC_STRIDE + i * SC_STRIDE + j];
        }
    }
    __syncthreads();

    // ---------------- Phase 5: attn_out = probs @ V  (B from transposed VT) ---------------
    {
        _Float16* PB = (_Float16*)(smem + OFF_PB);
        _Float16* AO = (_Float16*)(smem + OFF_AO);   // overlays dead Q
#pragma unroll
        for (int n2 = 0; n2 < 4; ++n2) {
            const int nt = ng * 4 + n2;
            const int h  = nt >> 1;                  // 16 cols per tile, 32 per head
            v8f acc = {};
#pragma unroll
            for (int kc = 0; kc < 2; ++kc) {
                v16h a  = load_frag_a(PB + h * NPAD * PB_STRIDE, mt, PB_STRIDE, kc * 32);
                v16h bm = load_frag_b(sVT, nt, VT_STRIDE, kc * 32);
                acc = wmma_f16(a, bm, acc);
            }
            const int c = nt * 16 + nl;
#pragma unroll
            for (int r = 0; r < 8; ++r)
                AO[(mt * 16 + r + hi8) * LW_STRIDE + c] = (_Float16)acc[r];
        }
    }
    __syncthreads();

    // ---------------- Phase 6: out = attn_out @ Wo^T + bo -> spatial [B,C,H,W] ------------
    {
        _Float16* AO = (_Float16*)(smem + OFF_AO);
#pragma unroll
        for (int n2 = 0; n2 < 4; ++n2) {
            const int nt = ng * 4 + n2;
            v8f acc = {};
#pragma unroll
            for (int kc = 0; kc < 4; ++kc) {
                v16h a  = load_frag_a(AO,  mt, LW_STRIDE, kc * 32);
                v16h bm = load_frag_b(sWo, nt, LW_STRIDE, kc * 32);
                acc = wmma_f16(a, bm, acc);
            }
            const int c = nt * 16 + nl;
            const float bias = bo[c];
#pragma unroll
            for (int r = 0; r < 8; ++r) {
                const int t = mt * 16 + r + hi8;
                if (t < N_TOK) {
                    const int yy = y0 + t / 7, xx = x0 + t % 7;
                    out_spatial[base + (c * HW + yy) * HW + xx] = acc[r] + bias;
                }
            }
        }
    }
}

extern "C" void kernel_launch(void* const* d_in, const int* in_sizes, int n_in,
                              void* d_out, int out_size, void* d_ws, size_t ws_size,
                              hipStream_t stream) {
    const float* q   = (const float*)d_in[0];
    const float* k   = (const float*)d_in[1];
    const float* v   = (const float*)d_in[2];
    const float* Wq  = (const float*)d_in[3];
    const float* bq  = (const float*)d_in[4];
    const float* Wk  = (const float*)d_in[5];
    const float* bk  = (const float*)d_in[6];
    const float* Wv  = (const float*)d_in[7];
    const float* bv  = (const float*)d_in[8];
    const float* Wo  = (const float*)d_in[9];
    const float* bo  = (const float*)d_in[10];
    const float* rpb = (const float*)d_in[11];

    float* out_sp  = (float*)d_out;
    float* out_ctx = out_sp + (size_t)64 * 128 * 56 * 56;   // spatial then ctx, flat

    _Float16* w16 = (_Float16*)d_ws;                        // 139264 B padded f16 weights

    (void)in_sizes; (void)n_in; (void)out_size; (void)ws_size;

    // one-time (per launch) weight conversion into workspace
    const int welems = 4 * 128 * LW_STRIDE;
    convert_weights_f16<<<(welems + 255) / 256, 256, 0, stream>>>(Wq, Wk, Wv, Wo, w16);

    hipFuncSetAttribute((const void*)swin_window_attn_fused,
                        hipFuncAttributeMaxDynamicSharedMemorySize, SMEM_BYTES);

    dim3 grid(4096), block(256);
    swin_window_attn_fused<<<grid, block, SMEM_BYTES, stream>>>(
        q, k, v, w16, bq, bk, bv, bo, rpb, out_sp, out_ctx);
}
